// MambaLayer_33200097198889
// MI455X (gfx1250) — matrix operand order repaired and checked by reading, not compiled, not run on hardware
//
#include <hip/hip_runtime.h>
#include <hip/hip_bf16.h>

// Dims from the reference
#define BATCH 2
#define DM    48      // D_MODEL
#define DI    96      // D_INNER
#define NST   16      // D_STATE
#define DTR   3       // DT_RANK
#define KDIR  4
#define GD    20
#define LL    8000    // GD^3
#define NPOS  (BATCH*LL)   // 16000
#define NC    35      // DT_RANK + 2*D_STATE

typedef __attribute__((ext_vector_type(16))) _Float16 v16h;
typedef __attribute__((ext_vector_type(8)))  _Float16 v8h;
typedef __attribute__((ext_vector_type(8)))  float    v8f;

// ---- gfx1250 WMMA f16 16x16x32 operand loads (CDNA5 ISA) ----
// A (16x32 f16): lane = half*16+m; elements 0..7 -> K=kb+half*8+0..7,
//                elements 8..15 -> K=kb+16+half*8+0..7  (two 16B blocks)
__device__ __forceinline__ v16h load_a(const _Float16* __restrict__ row, int kb, int half) {
  v8h lo = *(const v8h*)(row + kb + (half << 3));
  v8h hi = *(const v8h*)(row + kb + 16 + (half << 3));
  v16h r;
#pragma unroll
  for (int i = 0; i < 8; ++i) { r[i] = lo[i]; r[i + 8] = hi[i]; }
  return r;
}
// B (32x16 f16): lane = half*16+n; elements 0..7 -> K=kb+half*8+0..7,
//                elements 8..15 -> K=kb+16+half*8+0..7  (two 16B blocks)
__device__ __forceinline__ v16h load_b(const _Float16* __restrict__ row, int kb, int half) {
  v8h lo = *(const v8h*)(row + kb + (half << 3));
  v8h hi = *(const v8h*)(row + kb + 16 + (half << 3));
  v16h r;
#pragma unroll
  for (int i = 0; i < 8; ++i) { r[i] = lo[i]; r[i + 8] = hi[i]; }
  return r;
}

// Row map for _directions() along L (k=0 id, k=1 reversed, k=3 front/back interleave)
__device__ __forceinline__ int dir_row(int k, int l) {
  if (k == 1) return LL - 1 - l;
  if (k == 3) return (l & 1) ? (LL - 1 - (l >> 1)) : (l >> 1);
  return l;
}

// ---------------- elementwise / stencil kernels ----------------

// x (B,L,C) -> u0 (B,C,L)
__global__ void k_transpose(const float* __restrict__ x, float* __restrict__ u0) {
  int i = blockIdx.x * blockDim.x + threadIdx.x;
  if (i >= BATCH * DM * LL) return;
  int l = i % LL; int t = i / LL; int c = t % DM; int b = t / DM;
  u0[i] = x[((size_t)b * LL + l) * DM + c];
}

// one step of u += tau * Lap(u), zero-padded 7-point stencil
__global__ void k_lap(const float* __restrict__ uin, float* __restrict__ uout) {
  int i = blockIdx.x * blockDim.x + threadIdx.x;
  if (i >= BATCH * DM * LL) return;
  int l = i % LL; int bc_ = i / LL;
  int ww = l % GD, hh = (l / GD) % GD, dd = l / (GD * GD);
  const float* base = uin + (size_t)bc_ * LL;
  float c = base[l];
  float acc = -6.f * c;
  if (dd > 0)      acc += base[l - GD * GD];
  if (dd < GD - 1) acc += base[l + GD * GD];
  if (hh > 0)      acc += base[l - GD];
  if (hh < GD - 1) acc += base[l + GD];
  if (ww > 0)      acc += base[l - 1];
  if (ww < GD - 1) acc += base[l + 1];
  uout[i] = c + 0.1f * acc;
}

// Pre-compose Wch = f16(in_proj_cross_w (96x48) @ concat_w (48x96)), bc = W2 @ concat_b
__global__ void k_wc(const float* __restrict__ w2, const float* __restrict__ w1,
                     const float* __restrict__ b1, _Float16* __restrict__ Wch,
                     float* __restrict__ bc) {
  int i = blockIdx.x * blockDim.x + threadIdx.x;
  if (i < DI * DI) {
    int d = i / DI, c = i % DI;
    float s = 0.f;
    for (int o = 0; o < DM; ++o) s += w2[d * DM + o] * w1[o * DI + c];
    Wch[i] = (_Float16)s;
  } else if (i < DI * DI + DI) {
    int d = i - DI * DI;
    float s = 0.f;
    for (int o = 0; o < DM; ++o) s += w2[d * DM + o] * b1[o];
    bc[d] = s;
  }
}

// Convert all static weights to padded f16 row-major buffers
__global__ void k_wcvt(const float* __restrict__ inproj_w, const float* __restrict__ xpw,
                       const float* __restrict__ outp_w, _Float16* __restrict__ Wih,
                       _Float16* __restrict__ Xpwh, _Float16* __restrict__ Woh) {
  int i = blockIdx.x * blockDim.x + threadIdx.x;
  const int N1 = 192 * 64, N2 = KDIR * 48 * 96, N3 = 48 * 96;
  if (i < N1) {                               // in_proj (192,48) -> (192,64) padded
    int o = i >> 6, c = i & 63;
    Wih[i] = (_Float16)((c < DM) ? inproj_w[o * DM + c] : 0.f);
  } else if (i < N1 + N2) {                   // x_proj (4,35,96) -> (4,48,96) padded rows
    int j = i - N1; int d = j % DI; int t = j / DI; int c = t % 48; int kd = t / 48;
    Xpwh[j] = (_Float16)((c < NC) ? xpw[((size_t)kd * NC + c) * DI + d] : 0.f);
  } else if (i < N1 + N2 + N3) {              // out_proj (48,96)
    int j = i - N1 - N2;
    Woh[j] = (_Float16)outp_w[j];
  }
}

// x padded to f16 (B*L, 64)
__global__ void k_cvt_x(const float* __restrict__ x, _Float16* __restrict__ xh) {
  int i = blockIdx.x * blockDim.x + threadIdx.x;
  if (i >= NPOS * 64) return;
  int c = i & 63; int m = i >> 6;
  xh[i] = (_Float16)((c < DM) ? x[(size_t)m * DM + c] : 0.f);
}

// cat(low, high) as f16 position-major (B*L, 96)
__global__ void k_cat(const float* __restrict__ x, const float* __restrict__ u0,
                      _Float16* __restrict__ cath) {
  int i = blockIdx.x * blockDim.x + threadIdx.x;
  if (i >= NPOS * DI) return;
  int c = i % DI; int m = i / DI; int b = m / LL, l = m % LL;
  float v;
  if (c < DM) v = u0[((size_t)b * DM + c) * LL + l];
  else {
    int cc = c - DM;
    v = x[(size_t)m * DM + cc] - u0[((size_t)b * DM + cc) * LL + l];
  }
  cath[i] = (_Float16)v;
}

// depthwise 3x3x3 conv + SiLU; input channel-major (B,96,L), output position-major (B,L,96)
__global__ void k_dwconv(const float* __restrict__ xi, const float* __restrict__ w,
                         const float* __restrict__ bias, float* __restrict__ xict) {
  int i = blockIdx.x * blockDim.x + threadIdx.x;
  if (i >= BATCH * DI * LL) return;
  int l = i % LL; int t = i / LL; int d = t % DI; int b = t / DI;
  int ww = l % GD, hh = (l / GD) % GD, dd = l / (GD * GD);
  const float* base = xi + (size_t)t * LL;
  float s = bias[d];
  for (int kd = 0; kd < 3; ++kd) {
    int zd = dd + kd - 1; if (zd < 0 || zd >= GD) continue;
    for (int kh = 0; kh < 3; ++kh) {
      int zh = hh + kh - 1; if (zh < 0 || zh >= GD) continue;
      for (int kw = 0; kw < 3; ++kw) {
        int zw = ww + kw - 1; if (zw < 0 || zw >= GD) continue;
        s += w[d * 27 + kd * 9 + kh * 3 + kw] * base[(zd * GD + zh) * GD + zw];
      }
    }
  }
  xict[((size_t)b * LL + l) * DI + d] = s / (1.f + __expf(-s));   // SiLU
}

// delta = softplus(dt_proj(dts) + bias), POSITION-major output [bk][l][96]
__global__ void k_delta(const float* __restrict__ dts, const float* __restrict__ dtw,
                        const float* __restrict__ dtb, float* __restrict__ deltat) {
  int i = blockIdx.x * blockDim.x + threadIdx.x;
  if (i >= BATCH * KDIR * LL * DI) return;
  int d = i % DI; int t = i / DI; int l = t % LL; int bk = t / LL;
  int kd = bk & 3;
  float s = dtb[kd * DI + d];
  const float* xb = dts + ((size_t)bk * DTR) * LL + l;
  for (int r = 0; r < DTR; ++r)
    s += xb[(size_t)r * LL] * dtw[((size_t)kd * DI + d) * DTR + r];
  deltat[i] = (s > 20.f) ? s : log1pf(__expf(s));
}

__global__ void k_zero(float* __restrict__ p, int n) {
  int i = blockIdx.x * blockDim.x + threadIdx.x;
  if (i < n) p[i] = 0.f;
}

// ---------------- WMMA GEMM kernels (one 16x16 tile per wave) ----------------

// xz = x @ in_proj_w.T ; ntile<6 -> xi channel-major (B,96,L), else -> z (B*L,96)
__global__ void k_inproj_gemm(const _Float16* __restrict__ xh, const _Float16* __restrict__ Wih,
                              float* __restrict__ xi, float* __restrict__ z) {
  int wave = threadIdx.x >> 5, lane = threadIdx.x & 31;
  int mtile = blockIdx.x * 8 + wave;   // 1000 m-tiles
  int ntile = blockIdx.y;              // 12 n-tiles (N=192)
  int half = lane >> 4, mr = lane & 15;
  int m0 = mtile * 16;
  const _Float16* arow = xh + (size_t)(m0 + mr) * 64;
  const _Float16* brow = Wih + (size_t)(ntile * 16 + mr) * 64;
  v8f acc = {};
#pragma unroll
  for (int kc = 0; kc < 2; ++kc) {     // K=48 padded to 64
    v16h a = load_a(arow, kc * 32, half);
    v16h bm = load_b(brow, kc * 32, half);
    acc = __builtin_amdgcn_wmma_f32_16x16x32_f16(false, a, false, bm, (short)0, acc, false, false);
  }
  int b = m0 / LL;                     // tile never straddles batch (LL % 16 == 0)
  int l0 = m0 - b * LL + (half << 3);
  if (ntile < 6) {                     // block-uniform split
    int n = ntile * 16 + mr;
    float* dst = xi + ((size_t)b * DI + n) * LL + l0;
#pragma unroll
    for (int r = 0; r < 8; ++r) dst[r] = acc[r];
  } else {
    int n = ntile * 16 + mr - DI;
    float* dst = z + ((size_t)b * LL + l0) * DI + n;
#pragma unroll
    for (int r = 0; r < 8; ++r) dst[(size_t)r * DI] = acc[r];
  }
}

// xc = Wch @ cat + bc; emit f16 position-major plus channel-reversed copy
__global__ void k_xc_gemm(const _Float16* __restrict__ cath, const _Float16* __restrict__ Wch,
                          const float* __restrict__ bc, _Float16* __restrict__ xch_pos,
                          _Float16* __restrict__ xch_rev) {
  int wave = threadIdx.x >> 5, lane = threadIdx.x & 31;
  int mtile = blockIdx.x * 8 + wave;   // 1000
  int ntile = blockIdx.y;              // 6 (N=96)
  int half = lane >> 4, mr = lane & 15;
  int m0 = mtile * 16;
  const _Float16* arow = cath + (size_t)(m0 + mr) * DI;
  const _Float16* brow = Wch + (size_t)(ntile * 16 + mr) * DI;
  v8f acc = {};
#pragma unroll
  for (int kc = 0; kc < 3; ++kc) {     // K=96
    v16h a = load_a(arow, kc * 32, half);
    v16h bm = load_b(brow, kc * 32, half);
    acc = __builtin_amdgcn_wmma_f32_16x16x32_f16(false, a, false, bm, (short)0, acc, false, false);
  }
  int n = ntile * 16 + mr;
  float bias = bc[n];
  int mbase = m0 + (half << 3);
  _Float16* p0 = xch_pos + (size_t)mbase * DI + n;
  _Float16* p1 = xch_rev + (size_t)mbase * DI + (DI - 1 - n);
#pragma unroll
  for (int r = 0; r < 8; ++r) {
    _Float16 v = (_Float16)(acc[r] + bias);
    p0[(size_t)r * DI] = v;
    p1[(size_t)r * DI] = v;
  }
}

// x_dbl: c<3 -> dts planar [bk][3][L]; 3<=c<35 -> bct [bk][l][32] (B_t|C_t contiguous)
__global__ void k_xdbl_gemm(const _Float16* __restrict__ xch_pos,
                            const _Float16* __restrict__ xch_rev,
                            const _Float16* __restrict__ Xpwh,
                            float* __restrict__ dts, float* __restrict__ bct) {
  int wave = threadIdx.x >> 5, lane = threadIdx.x & 31;
  int mtile = blockIdx.x * 8 + wave;
  if (mtile >= LL / 16) return;        // wave-uniform guard (500 l-tiles)
  int ntile = blockIdx.y;              // 3 (N padded 35->48)
  int bk = blockIdx.z; int b = bk >> 2, kd = bk & 3;
  int half = lane >> 4, mr = lane & 15;
  int lrow = mtile * 16 + mr;
  int sl = dir_row(kd, lrow);
  const _Float16* src = (kd == 2) ? xch_rev : xch_pos;
  const _Float16* arow = src + ((size_t)b * LL + sl) * DI;
  const _Float16* brow = Xpwh + ((size_t)kd * 48 + ntile * 16 + mr) * DI;
  v8f acc = {};
#pragma unroll
  for (int kc = 0; kc < 3; ++kc) {     // K=96 (d)
    v16h a = load_a(arow, kc * 32, half);
    v16h bm = load_b(brow, kc * 32, half);
    acc = __builtin_amdgcn_wmma_f32_16x16x32_f16(false, a, false, bm, (short)0, acc, false, false);
  }
  int c = ntile * 16 + mr;
  int lb = mtile * 16 + (half << 3);
  if (c >= DTR && c < NC) {            // B/C rows, contiguous-per-step layout
    float* dst = bct + ((size_t)bk * LL + lb) * 32 + (c - DTR);
#pragma unroll
    for (int r = 0; r < 8; ++r) dst[(size_t)r * 32] = acc[r];
  } else if (c < DTR) {                // dt-rank rows, planar
    float* dst = dts + ((size_t)bk * DTR + c) * LL + lb;
#pragma unroll
    for (int r = 0; r < 8; ++r) dst[r] = acc[r];
  }
}

// out = g @ out_proj_w.T  (M=16000, K=96, N=48)
__global__ void k_out_gemm(const _Float16* __restrict__ gh, const _Float16* __restrict__ Woh,
                           float* __restrict__ out) {
  int wave = threadIdx.x >> 5, lane = threadIdx.x & 31;
  int mtile = blockIdx.x * 8 + wave;   // 1000
  int ntile = blockIdx.y;              // 3
  int half = lane >> 4, mr = lane & 15;
  int m0 = mtile * 16;
  const _Float16* arow = gh + (size_t)(m0 + mr) * DI;
  const _Float16* brow = Woh + (size_t)(ntile * 16 + mr) * DI;
  v8f acc = {};
#pragma unroll
  for (int kc = 0; kc < 3; ++kc) {
    v16h a = load_a(arow, kc * 32, half);
    v16h bm = load_b(brow, kc * 32, half);
    acc = __builtin_amdgcn_wmma_f32_16x16x32_f16(false, a, false, bm, (short)0, acc, false, false);
  }
  float* dst = out + (size_t)(m0 + (half << 3)) * DM + ntile * 16 + mr;
#pragma unroll
  for (int r = 0; r < 8; ++r) dst[(size_t)r * DM] = acc[r];
}

// ---------------- selective scan ----------------
// One workgroup per (b,direction). Lanes 0..95: one channel each, 16-state
// recurrence in registers; all per-step operands are row-contiguous.
// Wave 3 (lanes 96..127) double-buffers next step's B/C through LDS.
// Compute lanes prefetch delta/u rows 8 steps ahead (global_prefetch_b8).
__global__ void k_scan(const float* __restrict__ deltat, const float* __restrict__ xict,
                       const float* __restrict__ bct, const float* __restrict__ A_logs,
                       const float* __restrict__ Dsv, float* __restrict__ ymean) {
  int bk = blockIdx.x; int b = bk >> 2, kd = bk & 3;
  int tid = threadIdx.x;
  __shared__ float sBC[2][32];
  float h[NST], Ar[NST], dsval = 0.f;
  int d = tid;
  if (d < DI) {
#pragma unroll
    for (int n = 0; n < NST; ++n) {
      h[n] = 0.f;
      Ar[n] = -__expf(A_logs[((size_t)kd * DI + d) * NST + n]);
    }
    dsval = Dsv[kd * DI + d];
  }
  int sd = (kd == 2) ? (DI - 1 - d) : d;            // channel map for u gather
  const float* bcb = bct + (size_t)bk * LL * 32;
  const float* drow = deltat + (size_t)bk * LL * DI + (d < DI ? d : 0);
  const float* urow = xict + (size_t)b * LL * DI + (d < DI ? sd : 0);
  float* yrow = ymean + (size_t)b * LL * DI + (d < DI ? d : 0);
  if (tid >= 96) sBC[0][tid - 96] = bcb[tid - 96];  // prime step 0
  __syncthreads();
  for (int l = 0; l < LL; ++l) {
    int cur = l & 1;
    if (tid >= 96) {
      int ln = l + 1;
      if (ln < LL) sBC[cur ^ 1][tid - 96] = bcb[(size_t)ln * 32 + (tid - 96)];
    } else {
      float dl = drow[(size_t)l * DI];
      int sl = dir_row(kd, l);
      float u = urow[(size_t)sl * DI];
      if (l + 8 < LL) {
        __builtin_prefetch(&drow[(size_t)(l + 8) * DI], 0, 0);
        __builtin_prefetch(&urow[(size_t)dir_row(kd, l + 8) * DI], 0, 0);
      }
      float du = dl * u;
      float y = 0.f;
#pragma unroll
      for (int n = 0; n < NST; ++n) {
        h[n] = __expf(dl * Ar[n]) * h[n] + du * sBC[cur][n];
        y += h[n] * sBC[cur][NST + n];
      }
      atomicAdd(&yrow[(size_t)l * DI], 0.25f * (y + dsval * u));
    }
    __syncthreads();
  }
}

// LayerNorm over 96 channels + sigmoid gate -> gh (B*L, 96) f16
__global__ void k_ln_gate(const float* __restrict__ ymean, const float* __restrict__ w,
                          const float* __restrict__ bb, const float* __restrict__ z,
                          _Float16* __restrict__ gh) {
  int m = blockIdx.x * blockDim.x + threadIdx.x;
  if (m >= NPOS) return;
  const float* yp = ymean + (size_t)m * DI;
  float s = 0.f, sq = 0.f;
  for (int i = 0; i < DI; ++i) { float v = yp[i]; s += v; sq += v * v; }
  float mu = s / DI;
  float var = sq / DI - mu * mu;
  float rstd = rsqrtf(var + 1e-5f);
  const float* zp = z + (size_t)m * DI;
  _Float16* gp = gh + (size_t)m * DI;
  for (int i = 0; i < DI; ++i) {
    float v = (yp[i] - mu) * rstd * w[i] + bb[i];
    gp[i] = (_Float16)(v * (1.f / (1.f + __expf(-zp[i]))));
  }
}

// ---------------- host launcher ----------------

extern "C" void kernel_launch(void* const* d_in, const int* in_sizes, int n_in,
                              void* d_out, int out_size, void* d_ws, size_t ws_size,
                              hipStream_t stream) {
  const float* x        = (const float*)d_in[0];
  const float* inproj_w = (const float*)d_in[1];
  const float* cross_w  = (const float*)d_in[2];
  const float* conv_w   = (const float*)d_in[3];
  const float* conv_b   = (const float*)d_in[4];
  const float* xproj_w  = (const float*)d_in[5];
  const float* dtw      = (const float*)d_in[6];
  const float* dtb      = (const float*)d_in[7];
  const float* A_logs   = (const float*)d_in[8];
  const float* Dsv      = (const float*)d_in[9];
  const float* onw      = (const float*)d_in[10];
  const float* onb      = (const float*)d_in[11];
  const float* outp_w   = (const float*)d_in[12];
  const float* concat_w = (const float*)d_in[13];
  const float* concat_b = (const float*)d_in[14];
  float* out = (float*)d_out;

  // f32 region
  float* ws = (float*)d_ws;
  float* u0     = ws;
  float* u1     = u0     + (size_t)BATCH * DM * LL;
  float* xi     = u1     + (size_t)BATCH * DM * LL;
  float* zb     = xi     + (size_t)BATCH * DI * LL;
  float* xict   = zb     + (size_t)NPOS * DI;
  float* dts    = xict   + (size_t)NPOS * DI;
  float* bct    = dts    + (size_t)BATCH * KDIR * DTR * LL;
  float* deltat = bct    + (size_t)BATCH * KDIR * LL * 32;
  float* ymean  = deltat + (size_t)BATCH * KDIR * LL * DI;
  float* bc     = ymean  + (size_t)NPOS * DI;
  // f16 region (32B-aligned: cumulative f32 bytes are a multiple of 32)
  _Float16* xh      = (_Float16*)(bc + 96);
  _Float16* cath    = xh      + (size_t)NPOS * 64;
  _Float16* xch_pos = cath    + (size_t)NPOS * DI;
  _Float16* xch_rev = xch_pos + (size_t)NPOS * DI;
  _Float16* gh      = xch_rev + (size_t)NPOS * DI;
  _Float16* Wch     = gh      + (size_t)NPOS * DI;
  _Float16* Wih     = Wch     + DI * DI;
  _Float16* Xpwh    = Wih     + 192 * 64;
  _Float16* Woh     = Xpwh    + (size_t)KDIR * 48 * DI;

  const int TB = 256;
  // 1) transpose x -> channel-major u0 ; convert x to padded f16
  k_transpose<<<(BATCH * DM * LL + TB - 1) / TB, TB, 0, stream>>>(x, u0);
  k_cvt_x<<<(NPOS * 64 + TB - 1) / TB, TB, 0, stream>>>(x, xh);
  // 2) weight prep (f16, padded)
  k_wcvt<<<(192 * 64 + KDIR * 48 * 96 + 48 * 96 + TB - 1) / TB, TB, 0, stream>>>(
      inproj_w, xproj_w, outp_w, Wih, Xpwh, Woh);
  k_wc<<<(DI * DI + DI + TB - 1) / TB, TB, 0, stream>>>(cross_w, concat_w, concat_b, Wch, bc);
  // 3) 10 Laplacian smoothing steps (ping-pong, result lands back in u0)
  for (int it = 0; it < 10; ++it) {
    const float* src = (it & 1) ? u1 : u0;
    float* dst       = (it & 1) ? u0 : u1;
    k_lap<<<(BATCH * DM * LL + TB - 1) / TB, TB, 0, stream>>>(src, dst);
  }
  // 4) cat(low, high) in f16 position-major
  k_cat<<<(NPOS * DI + TB - 1) / TB, TB, 0, stream>>>(x, u0, cath);
  // 5) xc = Wch @ cat + bc -> f16 (+ channel-reversed copy)   [WMMA]
  k_xc_gemm<<<dim3(125, 6), TB, 0, stream>>>(cath, Wch, bc, xch_pos, xch_rev);
  // 6) xz = x @ in_proj_w.T -> xi, z   [WMMA]
  k_inproj_gemm<<<dim3(125, 12), TB, 0, stream>>>(xh, Wih, xi, zb);
  // 7) depthwise conv + SiLU -> u position-major
  k_dwconv<<<(BATCH * DI * LL + TB - 1) / TB, TB, 0, stream>>>(xi, conv_w, conv_b, xict);
  // 8) x_dbl per-direction projection -> dts planar + bct [bk][l][32]   [WMMA]
  k_xdbl_gemm<<<dim3(63, 3, BATCH * KDIR), TB, 0, stream>>>(xch_pos, xch_rev, Xpwh, dts, bct);
  // 9) delta = softplus(dt projection), position-major
  k_delta<<<(BATCH * KDIR * LL * DI + TB - 1) / TB, TB, 0, stream>>>(dts, dtw, dtb, deltat);
  // 10) zero the direction-mean accumulator
  k_zero<<<(NPOS * DI + TB - 1) / TB, TB, 0, stream>>>(ymean, NPOS * DI);
  // 11) selective scan (8000 steps), pipelined B/C via LDS, mean via atomics
  k_scan<<<BATCH * KDIR, 128, 0, stream>>>(deltat, xict, bct, A_logs, Dsv, ymean);
  // 12) LayerNorm + sigmoid(z) gate -> f16 g
  k_ln_gate<<<(NPOS + TB - 1) / TB, TB, 0, stream>>>(ymean, onw, onb, zb, gh);
  // 13) final projection to D_MODEL  [WMMA]
  k_out_gemm<<<dim3(125, 3), TB, 0, stream>>>(gh, Woh, out);
}
